// SeqToSeqTransformerModel_55405078119352
// MI455X (gfx1250) — matrix operand order repaired and checked by reading, not compiled, not run on hardware
//
#include <hip/hip_runtime.h>
#include <hip/hip_bf16.h>

// ---------------------------------------------------------------------------
// Seq2Seq transformer forward for MI455X (gfx1250, wave32, WMMA).
// All GEMMs use v_wmma_f32_16x16x32_f16 (f16 A/B, f32 accumulate).
// GEMM is explicitly software-pipelined (double-buffered fragments) so the
// global b128 loads overlap the WMMA pipe instead of s_wait_loadcnt 0 stalls.
// Weights converted+transposed to f16 once per call (~5 MB, L2-resident).
// Residual stream kept f32; bias/ReLU/residual fused into GEMM epilogue.
// Attention: one block per (batch, head); K / V^T / P staged in LDS.
// ---------------------------------------------------------------------------

typedef _Float16 h16;
typedef __attribute__((ext_vector_type(16))) _Float16 v16h;
typedef __attribute__((ext_vector_type(8)))  _Float16 v8h;
typedef __attribute__((ext_vector_type(4)))  _Float16 v4h;
typedef __attribute__((ext_vector_type(8)))  float    v8f;

#define BB   512
#define TT   128
#define CC   128
#define HH   4
#define HSZ  32
#define FFD  512
#define BT   (BB * TT)
#define VOUTD 256

// ---- WMMA fragment helpers (wave32; contiguous k order, same for A and B) ----------
// A (16x32 f16): lanes 0-15 row M=lane, K=0..15;
//                lanes 16-31 same rows, K=16..31.
__device__ __forceinline__ v16h load_afrag(const h16* __restrict__ base, int stride, int lane) {
  int r  = lane & 15;
  int kb = (lane >> 4) << 4;
  const h16* p = base + r * stride + kb;
  v8h lo = *(const v8h*)(p);
  v8h hi = *(const v8h*)(p + 8);
  return __builtin_shufflevector(lo, hi, 0,1,2,3,4,5,6,7,8,9,10,11,12,13,14,15);
}
// B (32x16 f16) from an N-major (transposed) buffer BT[n*strideK + k]:
// lanes 0-15: col n=lane, K=0..15 ; lanes 16-31: col n=lane-16, K=16..31.
__device__ __forceinline__ v16h load_bfrag(const h16* __restrict__ baseT, int strideK, int lane) {
  int n  = lane & 15;
  int kb = (lane >> 4) << 4;
  return *(const v16h*)(baseT + n * strideK + kb);
}
__device__ __forceinline__ v8f wmma_f16(v16h a, v16h b, v8f c) {
  return __builtin_amdgcn_wmma_f32_16x16x32_f16(false, a, false, b, (short)0, c, false, false);
}

// ---- embedding: out[b*T+t, c] = tok[idx[b,t], c] + pos[t, c] --------------
__global__ __launch_bounds__(256) void embed_kernel(
    const int* __restrict__ idx, const float* __restrict__ tok,
    const float* __restrict__ pos, float* __restrict__ out, int total) {
  int g = blockIdx.x * 256 + threadIdx.x;
  if (g >= total) return;
  int row = g >> 7;          // / CC
  int c   = g & 127;
  int t   = row & (TT - 1);  // row % T
  out[g] = tok[idx[row] * CC + c] + pos[t * CC + c];
}

// ---- LayerNorm over C=128, f32 in -> f16 out. One wave32 per row. ---------
__global__ __launch_bounds__(256) void ln_f16_kernel(
    const float* __restrict__ X, const float* __restrict__ g,
    const float* __restrict__ b, h16* __restrict__ Y, int rows) {
  int lane = threadIdx.x & 31;
  int row  = blockIdx.x * 8 + (threadIdx.x >> 5);
  if (row >= rows) return;
  float4 x4 = *(const float4*)(X + row * CC + lane * 4);
  float s = x4.x + x4.y + x4.z + x4.w;
  #pragma unroll
  for (int off = 16; off; off >>= 1) s += __shfl_xor(s, off, 32);
  float mean = s * (1.0f / 128.0f);
  float d0 = x4.x - mean, d1 = x4.y - mean, d2 = x4.z - mean, d3 = x4.w - mean;
  float v = d0*d0 + d1*d1 + d2*d2 + d3*d3;
  #pragma unroll
  for (int off = 16; off; off >>= 1) v += __shfl_xor(v, off, 32);
  float inv = rsqrtf(v * (1.0f / 128.0f) + 1e-5f);
  float4 g4 = *(const float4*)(g + lane * 4);
  float4 b4 = *(const float4*)(b + lane * 4);
  v4h y;
  y[0] = (h16)(d0 * inv * g4.x + b4.x);
  y[1] = (h16)(d1 * inv * g4.y + b4.y);
  y[2] = (h16)(d2 * inv * g4.z + b4.z);
  y[3] = (h16)(d3 * inv * g4.w + b4.w);
  *(v4h*)(Y + row * CC + lane * 4) = y;
}

// ---- generic WMMA GEMM: out[M,N] = A[M,K] * W[K,N] (+bias)(+ReLU)(+resid) -
// W supplied pre-transposed: WT[n*K + k]. Block tile 128x64, 8 waves, each
// wave owns a 16x64 strip (4 accumulators). K-loop step 32, double-buffered:
// iteration i issues loads for step i, then WMMAs for step i-1, so the ten
// b128 loads overlap the matrix pipe.
__global__ __launch_bounds__(256) void gemm_f16_wmma(
    const h16* __restrict__ A, int lda,
    const h16* __restrict__ WT,
    const float* __restrict__ bias,
    const float* __restrict__ residual,
    float* __restrict__ outF, h16* __restrict__ outH,
    int ldo, int K, int relu) {
  int lane = threadIdx.x & 31;
  int wave = threadIdx.x >> 5;
  int r0 = blockIdx.y * 128 + wave * 16;
  int c0 = blockIdx.x * 64;

  v8f zero = {0.f,0.f,0.f,0.f,0.f,0.f,0.f,0.f};
  v8f acc0 = zero, acc1 = zero, acc2 = zero, acc3 = zero;

  const h16* Arow  = A + r0 * lda;
  const h16* Wbase = WT + c0 * K;

  // prologue: fragments for k0 = 0
  v16h af  = load_afrag(Arow, lda, lane);
  v16h bf0 = load_bfrag(Wbase + 0 * 16 * K, K, lane);
  v16h bf1 = load_bfrag(Wbase + 1 * 16 * K, K, lane);
  v16h bf2 = load_bfrag(Wbase + 2 * 16 * K, K, lane);
  v16h bf3 = load_bfrag(Wbase + 3 * 16 * K, K, lane);

  for (int k0 = 32; k0 < K; k0 += 32) {
    // issue next step's loads first (overlap with WMMAs below)
    v16h afn = load_afrag(Arow + k0, lda, lane);
    v16h bn0 = load_bfrag(Wbase + 0 * 16 * K + k0, K, lane);
    v16h bn1 = load_bfrag(Wbase + 1 * 16 * K + k0, K, lane);
    v16h bn2 = load_bfrag(Wbase + 2 * 16 * K + k0, K, lane);
    v16h bn3 = load_bfrag(Wbase + 3 * 16 * K + k0, K, lane);
    acc0 = wmma_f16(af, bf0, acc0);
    acc1 = wmma_f16(af, bf1, acc1);
    acc2 = wmma_f16(af, bf2, acc2);
    acc3 = wmma_f16(af, bf3, acc3);
    af = afn; bf0 = bn0; bf1 = bn1; bf2 = bn2; bf3 = bn3;
  }
  // epilogue MACs for the last K step
  acc0 = wmma_f16(af, bf0, acc0);
  acc1 = wmma_f16(af, bf1, acc1);
  acc2 = wmma_f16(af, bf2, acc2);
  acc3 = wmma_f16(af, bf3, acc3);

  v8f accv[4] = {acc0, acc1, acc2, acc3};
  int cl = lane & 15;
  int rofs = (lane >> 4) << 3;
  #pragma unroll
  for (int nf = 0; nf < 4; ++nf) {
    int col = c0 + nf * 16 + cl;
    float bv = bias ? bias[col] : 0.0f;
    #pragma unroll
    for (int gi = 0; gi < 8; ++gi) {
      int row = r0 + gi + rofs;
      float v = accv[nf][gi] + bv;
      if (relu) v = v > 0.0f ? v : 0.0f;
      int o = row * ldo + col;
      if (residual) v += residual[o];
      if (outF) outF[o] = v;
      if (outH) outH[o] = (h16)v;
    }
  }
}

// ---- attention: one block per (b,h). Q,K,V are f16 slices of a packed
// activation buffer (head columns h*32..h*32+31, row stride ldqkv).
// S = QK^T * scale + mask ; softmax ; O = P V. All matmuls via WMMA.
__global__ __launch_bounds__(256) void attention_kernel(
    const h16* __restrict__ Q, const h16* __restrict__ Kp,
    const h16* __restrict__ Vp, int ldqkv,
    const int* __restrict__ keyIdx,
    h16* __restrict__ O, int ldo, int causal) {
  __shared__ __align__(32) h16 Ks[TT * HSZ];        // [s][d]   8 KB
  __shared__ __align__(32) h16 Vt[HSZ * TT];        // [d][s]   8 KB (transposed)
  __shared__ __align__(32) h16 Ps[8][16 * TT];      // per-wave P tile, 32 KB
  __shared__ float maskS[TT];

  int b = blockIdx.y, h = blockIdx.x;
  int tid = threadIdx.x, lane = tid & 31, wave = tid >> 5;
  int rowBase = b * TT;

  { // stage K (row-major) and V (transposed) into LDS
    int s   = tid >> 1;
    int seg = (tid & 1) * 16;
    const h16* kg = Kp + (rowBase + s) * ldqkv + h * HSZ;
    *(v8h*)&Ks[s * HSZ + seg]     = *(const v8h*)(kg + seg);
    *(v8h*)&Ks[s * HSZ + seg + 8] = *(const v8h*)(kg + seg + 8);
    const h16* vg = Vp + (rowBase + s) * ldqkv + h * HSZ;
    v8h v0 = *(const v8h*)(vg + seg);
    v8h v1 = *(const v8h*)(vg + seg + 8);
    #pragma unroll
    for (int d = 0; d < 8; ++d) {
      Vt[(seg + d) * TT + s]     = v0[d];
      Vt[(seg + d + 8) * TT + s] = v1[d];
    }
  }
  if (tid < TT) maskS[tid] = (keyIdx[b * TT + tid] != 0) ? 0.0f : -1e30f;
  __syncthreads();

  int r0 = wave * 16;
  const h16* qg = Q + (rowBase + r0) * ldqkv + h * HSZ;
  v16h qa = load_afrag(qg, ldqkv, lane);   // K-dim = 32 = HS: one WMMA per tile

  v8f zero = {0.f,0.f,0.f,0.f,0.f,0.f,0.f,0.f};
  v8f sacc[8];
  // batch the LDS B-fragment loads in groups of 4 ahead of their WMMAs
  #pragma unroll
  for (int grp = 0; grp < 2; ++grp) {
    v16h kb0 = load_bfrag(&Ks[(grp * 4 + 0) * 16 * HSZ], HSZ, lane);
    v16h kb1 = load_bfrag(&Ks[(grp * 4 + 1) * 16 * HSZ], HSZ, lane);
    v16h kb2 = load_bfrag(&Ks[(grp * 4 + 2) * 16 * HSZ], HSZ, lane);
    v16h kb3 = load_bfrag(&Ks[(grp * 4 + 3) * 16 * HSZ], HSZ, lane);
    sacc[grp * 4 + 0] = wmma_f16(qa, kb0, zero);
    sacc[grp * 4 + 1] = wmma_f16(qa, kb1, zero);
    sacc[grp * 4 + 2] = wmma_f16(qa, kb2, zero);
    sacc[grp * 4 + 3] = wmma_f16(qa, kb3, zero);
  }

  // scale + masks + row softmax (row r = r0 + g + 8*(lane>=16), col = nt*16 + (lane&15))
  const float scale = 0.17677669529663687f;  // HS^-0.5
  int cl = lane & 15;
  int rofs = (lane >> 4) << 3;
  float mx[8], sum[8];
  #pragma unroll
  for (int gi = 0; gi < 8; ++gi) { mx[gi] = -3.4e38f; sum[gi] = 0.0f; }
  #pragma unroll
  for (int nt = 0; nt < 8; ++nt) {
    int col = nt * 16 + cl;
    float madd = maskS[col];
    #pragma unroll
    for (int gi = 0; gi < 8; ++gi) {
      float v = sacc[nt][gi] * scale + madd;
      if (causal && col > (r0 + gi + rofs)) v = -1e30f;
      sacc[nt][gi] = v;
      mx[gi] = fmaxf(mx[gi], v);
    }
  }
  #pragma unroll
  for (int gi = 0; gi < 8; ++gi)
    #pragma unroll
    for (int off = 8; off; off >>= 1) mx[gi] = fmaxf(mx[gi], __shfl_xor(mx[gi], off, 32));
  #pragma unroll
  for (int nt = 0; nt < 8; ++nt)
    #pragma unroll
    for (int gi = 0; gi < 8; ++gi) {
      float e = __expf(sacc[nt][gi] - mx[gi]);
      sacc[nt][gi] = e;
      sum[gi] += e;
    }
  #pragma unroll
  for (int gi = 0; gi < 8; ++gi) {
    #pragma unroll
    for (int off = 8; off; off >>= 1) sum[gi] += __shfl_xor(sum[gi], off, 32);
    sum[gi] = 1.0f / sum[gi];
  }
  // write normalized P tile (f16) to this wave's LDS strip, row-major [16][128]
  #pragma unroll
  for (int nt = 0; nt < 8; ++nt)
    #pragma unroll
    for (int gi = 0; gi < 8; ++gi)
      Ps[wave][(gi + rofs) * TT + nt * 16 + cl] = (h16)(sacc[nt][gi] * sum[gi]);
  __syncthreads();

  // O[16,32] = P[16,128] x V[128,32]
  v8f oacc[2] = {zero, zero};
  #pragma unroll
  for (int k0 = 0; k0 < TT; k0 += 32) {
    v16h pa = load_afrag(&Ps[wave][k0], TT, lane);
    v16h vb0 = load_bfrag(&Vt[0 * 16 * TT + k0], TT, lane);
    v16h vb1 = load_bfrag(&Vt[1 * 16 * TT + k0], TT, lane);
    oacc[0] = wmma_f16(pa, vb0, oacc[0]);
    oacc[1] = wmma_f16(pa, vb1, oacc[1]);
  }
  #pragma unroll
  for (int nf = 0; nf < 2; ++nf)
    #pragma unroll
    for (int gi = 0; gi < 8; ++gi) {
      int row = rowBase + r0 + gi + rofs;
      int col = h * HSZ + nf * 16 + cl;
      O[row * ldo + col] = (h16)oacc[nf][gi];
    }
}

// ---- weight conversion: f32 [K,N] -> f16 transposed WT[n*K + k] -----------
__global__ __launch_bounds__(256) void convT_kernel(
    const float* __restrict__ W, h16* __restrict__ WT, int K, int N) {
  int g = blockIdx.x * 256 + threadIdx.x;
  if (g >= K * N) return;
  int n = g / K, k = g - n * K;
  WT[g] = (h16)W[k * N + n];
}
// per-head weight [H=4][C=128][HS=32] -> WT[n*128 + c], n = h*32+d
__global__ __launch_bounds__(256) void convHeadT_kernel(
    const float* __restrict__ W, h16* __restrict__ WT) {
  int g = blockIdx.x * 256 + threadIdx.x;   // 16384 elements
  int n = g >> 7, c = g & 127;
  WT[g] = (h16)W[(n >> 5) * 4096 + c * 32 + (n & 31)];
}

// ---------------------------------------------------------------------------
extern "C" void kernel_launch(void* const* d_in, const int* in_sizes, int n_in,
                              void* d_out, int out_size, void* d_ws, size_t ws_size,
                              hipStream_t stream) {
  (void)in_sizes; (void)n_in; (void)out_size;
  auto F = [&](int i) { return (const float*)d_in[i]; };
  const int* input_idx  = (const int*)d_in[0];
  const int* output_idx = (const int*)d_in[1];

  // ---- workspace layout (bump allocator, 256B aligned) ----
  char* wsp = (char*)d_ws;
  size_t used = 0;
  auto alloc = [&](size_t bytes) {
    void* p = wsp + used;
    used += (bytes + 255) & ~(size_t)255;
    return p;
  };
  float* srcF  = (float*)alloc((size_t)BT * CC * 4);
  float* tgtF  = (float*)alloc((size_t)BT * CC * 4);
  h16*   lnA   = (h16*)  alloc((size_t)BT * CC * 2);
  h16*   lnB   = (h16*)  alloc((size_t)BT * CC * 2);
  h16*   qkv   = (h16*)  alloc((size_t)BT * 384 * 2);
  h16*   attnO = (h16*)  alloc((size_t)BT * CC * 2);
  h16*   ffnH  = (h16*)  alloc((size_t)BT * FFD * 2);
  h16*   wbuf  = (h16*)  alloc((size_t)2785280 * 2);   // all f16 weights
  if (used > ws_size) return;  // workspace too small; nothing safe to do

  size_t wofs = 0;
  auto allocW = [&](size_t halfs) { h16* p = wbuf + wofs; wofs += halfs; return p; };
  auto convQKV = [&](const float* Wq, const float* Wk, const float* Wv) {
    h16* p = allocW(3 * 128 * 128);
    convHeadT_kernel<<<64, 256, 0, stream>>>(Wq, p);
    convHeadT_kernel<<<64, 256, 0, stream>>>(Wk, p + 16384);
    convHeadT_kernel<<<64, 256, 0, stream>>>(Wv, p + 32768);
    return p;
  };
  auto convM = [&](const float* W, int K, int N) {
    h16* p = allocW((size_t)K * N);
    convT_kernel<<<(K * N + 255) / 256, 256, 0, stream>>>(W, p, K, N);
    return p;
  };
  auto gemm = [&](const h16* A, int lda, const h16* WT, const float* bias,
                  const float* resid, float* oF, h16* oH, int ldo, int N, int K, int relu) {
    dim3 grid(N / 64, BT / 128);
    gemm_f16_wmma<<<grid, 256, 0, stream>>>(A, lda, WT, bias, resid, oF, oH, ldo, K, relu);
  };
  auto ln = [&](const float* X, const float* g, const float* b, h16* Y) {
    ln_f16_kernel<<<BT / 8, 256, 0, stream>>>(X, g, b, Y, BT);
  };
  auto attn = [&](const h16* Qb, const h16* Kb, const h16* Vb, int ldq,
                  const int* kidx, h16* Ob, int causal) {
    attention_kernel<<<dim3(HH, BB), 256, 0, stream>>>(Qb, Kb, Vb, ldq, kidx, Ob, CC, causal);
  };

  // ---- embeddings ----
  embed_kernel<<<(BT * CC) / 256, 256, 0, stream>>>(input_idx,  F(2), F(4), srcF, BT * CC);
  embed_kernel<<<(BT * CC) / 256, 256, 0, stream>>>(output_idx, F(3), F(5), tgtF, BT * CC);

  // ---- encoder: layer l at index base 6 + 13*l ----
  for (int l = 0; l < 6; ++l) {
    int e = 6 + 13 * l;
    h16* Wqkv = convQKV(F(e + 2), F(e + 3), F(e + 4));
    h16* WpT  = convM(F(e + 5), 128, 128);
    h16* W1T  = convM(F(e + 9), 128, 512);
    h16* W2T  = convM(F(e + 11), 512, 128);

    ln(srcF, F(e + 0), F(e + 1), lnA);
    gemm(lnA, CC, Wqkv, nullptr, nullptr, nullptr, qkv, 384, 384, 128, 0);
    attn(qkv, qkv + 128, qkv + 256, 384, input_idx, attnO, 0);
    gemm(attnO, CC, WpT, F(e + 6), srcF, srcF, nullptr, CC, 128, 128, 0);
    ln(srcF, F(e + 7), F(e + 8), lnA);
    gemm(lnA, CC, W1T, F(e + 10), nullptr, nullptr, ffnH, FFD, 512, 128, 1);
    gemm(ffnH, FFD, W2T, F(e + 12), srcF, srcF, nullptr, CC, 128, 512, 0);
  }

  // ---- decoder: layer l at index base 84 + 22*l ----
  for (int l = 0; l < 6; ++l) {
    int d = 84 + 22 * l;
    h16* WqkvS = convQKV(F(d + 2), F(d + 3), F(d + 4));
    h16* WpTS  = convM(F(d + 5), 128, 128);
    h16* WqkvC = convQKV(F(d + 11), F(d + 12), F(d + 13));
    h16* WpTC  = convM(F(d + 14), 128, 128);
    h16* W1T   = convM(F(d + 18), 128, 512);
    h16* W2T   = convM(F(d + 20), 512, 128);

    // masked self-attention
    ln(tgtF, F(d + 0), F(d + 1), lnA);
    gemm(lnA, CC, WqkvS, nullptr, nullptr, nullptr, qkv, 384, 384, 128, 0);
    attn(qkv, qkv + 128, qkv + 256, 384, output_idx, attnO, 1);
    gemm(attnO, CC, WpTS, F(d + 6), tgtF, tgtF, nullptr, CC, 128, 128, 0);

    // cross-attention: Q from LN(tgt), K/V from LN(src), pad mask from input
    ln(srcF, F(d + 7), F(d + 8), lnB);
    ln(tgtF, F(d + 9), F(d + 10), lnA);
    gemm(lnA, CC, WqkvC,         nullptr, nullptr, nullptr, qkv,       384, 128, 128, 0);
    gemm(lnB, CC, WqkvC + 16384, nullptr, nullptr, nullptr, qkv + 128, 384, 256, 128, 0);
    attn(qkv, qkv + 128, qkv + 256, 384, input_idx, attnO, 0);
    gemm(attnO, CC, WpTC, F(d + 15), tgtF, tgtF, nullptr, CC, 128, 128, 0);

    // FFN
    ln(tgtF, F(d + 16), F(d + 17), lnA);
    gemm(lnA, CC, W1T, F(d + 19), nullptr, nullptr, ffnH, FFD, 512, 128, 1);
    gemm(ffnH, FFD, W2T, F(d + 21), tgtF, tgtF, nullptr, CC, 128, 512, 0);
  }

  // ---- final LN + vocab head -> f32 logits ----
  ln(tgtF, F(216), F(217), lnA);
  h16* WhT = convM(F(218), 128, VOUTD);
  gemm(lnA, CC, WhT, F(219), nullptr, (float*)d_out, nullptr, VOUTD, VOUTD, 128, 0);
}
